// aggregation_custom_12695923327642
// MI455X (gfx1250) — matrix operand order repaired and checked by reading, not compiled, not run on hardware
//
#include <hip/hip_runtime.h>

typedef __attribute__((ext_vector_type(16))) __bf16 v16bf;
typedef __attribute__((ext_vector_type(8)))  float  v8f;
typedef __attribute__((ext_vector_type(4)))  float  v4f;
typedef __attribute__((ext_vector_type(4)))  int    v4i;

union BF16x16 { v16bf v; __bf16 e[16]; };

__device__ __forceinline__ float clamp01(float x) {
  return fminf(fmaxf(x, 0.0f), 1.0f);
}

__global__ void zero_f32(float* __restrict__ p, int n) {
  int i = blockIdx.x * blockDim.x + threadIdx.x;
  int s = gridDim.x * blockDim.x;
  for (; i < n; i += s) p[i] = 0.0f;
}

// Per-tile body. FULL=true: all 8 edges valid -> no per-lane guards, no EXEC
// manipulation, vectorized b128 index load. FULL=false: tail tile (guarded).
template <bool FULL>
__device__ __forceinline__ void process_tile(
    const float* __restrict__ x, const int* __restrict__ index,
    float* __restrict__ out, int E, int ebase, int ln16, int half, float lp,
    const BF16x16 (&bfr)[8][2], float (*lds_tile)[132]) {
  // ---- Load A rows (row M = 2*edge + patch), stage fp32 x into LDS,
  //      pack bf16 A fragments per the 16-bit A 16x32 layout:
  //      a[j] (lane half h): K = s*32 + (j<8 ? 0 : 16) + h*8 + (j&7)
  const int eloc = ln16 >> 1;
  const int p    = ln16 & 1;
  int n = ebase + eloc;
  if (!FULL) { if (n >= E) n = E - 1; }  // clamp; garbage rows never scattered
  const float* xrow = x + n * 128 + p * 64;
  float* ldsrow = &lds_tile[eloc][p * 64];

  BF16x16 afr[2];
#pragma unroll
  for (int q = 0; q < 4; ++q) {
    const int off = q * 16 + half * 8;
    v4f lo = *(const v4f*)(xrow + off);
    v4f hi = *(const v4f*)(xrow + off + 4);
    *(v4f*)(ldsrow + off)     = lo;
    *(v4f*)(ldsrow + off + 4) = hi;
    const int s = q >> 1, piece = q & 1;
#pragma unroll
    for (int k = 0; k < 4; ++k) {
      afr[s].e[piece * 8 + k]     = (__bf16)lo[k];
      afr[s].e[piece * 8 + 4 + k] = (__bf16)hi[k];
    }
  }
  asm volatile("s_wait_dscnt 0x0" ::: "memory");  // intra-wave LDS RAW ordering

  // ---- Node indices for this lane's 4 edges (D-layout: lanes 0-15 hold
  //      rows M=0..7 -> edges 0..3; lanes 16-31 rows M=8..15 -> edges 4..7)
  const int eb = half * 4;
  int idx4[4];
  if (FULL) {
    v4i iv = *(const v4i*)(index + ebase + eb);  // 16B-aligned: ebase,eb mult of 4
#pragma unroll
    for (int i = 0; i < 4; ++i) idx4[i] = iv[i];
  } else {
#pragma unroll
    for (int i = 0; i < 4; ++i) {
      int ne = ebase + eb + i;
      idx4[i] = index[ne < E ? ne : 0];
    }
  }

#pragma unroll
  for (int t = 0; t < 4; ++t) {
    // gates for output channels o = 16t + col (chunk 0) and o = 64+16t+col (chunk 1)
    v8f accA = {};
    accA = __builtin_amdgcn_wmma_f32_16x16x32_bf16(false, afr[0].v, false, bfr[t][0].v,     (short)0, accA, false, false);
    accA = __builtin_amdgcn_wmma_f32_16x16x32_bf16(false, afr[1].v, false, bfr[t][1].v,     (short)0, accA, false, false);
    v8f accB = {};
    accB = __builtin_amdgcn_wmma_f32_16x16x32_bf16(false, afr[0].v, false, bfr[t + 4][0].v, (short)0, accB, false, false);
    accB = __builtin_amdgcn_wmma_f32_16x16x32_bf16(false, afr[1].v, false, bfr[t + 4][1].v, (short)0, accB, false, false);

    const int dcol = t * 16 + ln16;  // d in [0,64)
#pragma unroll
    for (int i = 0; i < 4; ++i) {
      const int e = eb + i;
      if (FULL || (ebase + e < E)) {
        const float e10 = lds_tile[e][dcol];       // e1[n, 0, d]
        const float e11 = lds_tile[e][64 + dcol];  // e1[n, 1, d]
        const int v = 2 * i;                       // acc elems: M=2e(+8h), patch 0/1
        const float g00 = clamp01(accA[v]);
        const float g01 = clamp01(accA[v + 1]);
        const float g10 = clamp01(accB[v]);
        const float g11 = clamp01(accB[v + 1]);
        float* ob = out + idx4[i] * 128;
        atomicAdd(ob + dcol,      lp * (e10 * g00 + e11 * g01) + e10);
        atomicAdd(ob + 64 + dcol, lp * (e10 * g10 + e11 * g11) + e11);
      }
    }
  }
}

// One wave processes 8 edges (16 edge-patch rows) per tile:
//   gates = clamp(e1 @ W^T, 0, 1) via v_wmma_f32_16x16x32_bf16
//   combined = |lp| * (sum_p e1[:,p,d] * g[:,p,c*64+d]) + x
//   atomicAdd-scatter combined into out[index[n], :]
__global__ __launch_bounds__(256)
void gated_scatter(const float* __restrict__ x,
                   const float* __restrict__ W,
                   const float* __restrict__ lparam,
                   const int*   __restrict__ index,
                   float* __restrict__ out,
                   int E, int numTiles) {
  __shared__ float lds_x[8][8][132];  // [wave][edge][feat, padded stride 132]
  const int lane = threadIdx.x & 31;
  const int wid  = threadIdx.x >> 5;
  const int ln16 = lane & 15;
  const int half = lane >> 4;
  const float lp = fabsf(lparam[0]);

  // ---- Build B fragments (W^T 32x16 bf16 tiles), loop-invariant, in registers.
  // B layout: lane holds column n = lane%16; K = (j<8 ? 0 : 16) + (lane/16)*8 + (j&7) within a K-step.
  BF16x16 bfr[8][2];
#pragma unroll
  for (int t = 0; t < 8; ++t) {
    const float* wrow = W + (t * 16 + ln16) * 64;
#pragma unroll
    for (int s = 0; s < 2; ++s) {
      const float* wp = wrow + s * 32 + half * 8;
#pragma unroll
      for (int j = 0; j < 16; ++j) bfr[t][s].e[j] = (__bf16)wp[(j & 7) + (j >> 3) * 16];
    }
  }

  const int gw = blockIdx.x * 8 + wid;
  const int gstride = gridDim.x * 8;
  for (int tile = gw; tile < numTiles; tile += gstride) {
    const int ebase = tile * 8;
    if (ebase + 8 <= E) {   // wave-uniform branch: EXEC stays all-ones
      process_tile<true >(x, index, out, E, ebase, ln16, half, lp, bfr, lds_x[wid]);
    } else {
      process_tile<false>(x, index, out, E, ebase, ln16, half, lp, bfr, lds_x[wid]);
    }
  }
}

extern "C" void kernel_launch(void* const* d_in, const int* in_sizes, int n_in,
                              void* d_out, int out_size, void* d_ws, size_t ws_size,
                              hipStream_t stream) {
  const float* x      = (const float*)d_in[0];
  const float* W      = (const float*)d_in[1];
  const float* lparam = (const float*)d_in[2];
  const int*   index  = (const int*)d_in[3];
  float* out = (float*)d_out;

  const int E = in_sizes[3];  // number of edges
  const int numTiles = (E + 7) / 8;

  zero_f32<<<dim3(1024), dim3(256), 0, stream>>>(out, out_size);

  int blocks = (numTiles + 7) / 8;  // 8 waves per block
  if (blocks > 4096) blocks = 4096;
  if (blocks < 1) blocks = 1;
  gated_scatter<<<dim3(blocks), dim3(256), 0, stream>>>(x, W, lparam, index, out, E, numTiles);
}
